// Decoder_54692113547912
// MI455X (gfx1250) — matrix operand order repaired and checked by reading, not compiled, not run on hardware
//
#include <hip/hip_runtime.h>
#include <hip/hip_bf16.h>

// ---------------------------------------------------------------------------
// Attention + 2-layer LSTM decoder, B=16 T=256 H=512 V=32.
//  - Precompute (parallel, WMMA bf16): Kproj = enc@Wk^T+bk ; Xe = enc@WihE^T+b
//  - Persistent 16-WG kernel scans T steps; per-step GEMMs are
//    v_wmma_f32_16x16x32_bf16 tiles (M=16 == batch). Grid sync = atomic
//    sense barrier + s_cluster_barrier fast path. Working set is L2-resident.
// ---------------------------------------------------------------------------

typedef __attribute__((ext_vector_type(16))) __bf16 v16bf;
typedef __attribute__((ext_vector_type(8)))  float  v8f;

#define B_  16
#define T_  256
#define H_  512
#define V_  32
#define G_  2048   // 4*H
#define K2_ 1024   // 2*H
#define NWG 16

// ---- WMMA fragment helpers (contiguous k order k = 16h + s, wave32) --------------------
// A 16x32 bf16: lane m=l&15; lanes<16 hold K 0..15, lanes>=16 hold
// K 16..31 -> two contiguous 16B groups at khalf and khalf+8.
static __device__ inline v16bf loadA_frag(const __bf16* p, int khalf) {
  union { uint4 q[2]; v16bf v; } u;
  u.q[0] = *(const uint4*)(p + khalf);
  u.q[1] = *(const uint4*)(p + khalf + 8);
  return u.v;
}
// B 32x16 bf16 from W row-major [N][K] (B[k][n]=W[n][k]): lane n=l&15,
// lanes<16 -> K 0..15, lanes>=16 -> K 16..31 : 16 contiguous bf16.
static __device__ inline v16bf loadB_frag(const __bf16* p) {
  union { uint4 q[2]; v16bf v; } u;
  u.q[0] = *(const uint4*)(p);
  u.q[1] = *(const uint4*)(p + 8);
  return u.v;
}
static __device__ inline v8f wmma_bf16(v16bf a, v16bf b, v8f c) {
  return __builtin_amdgcn_wmma_f32_16x16x32_bf16(false, a, false, b,
                                                 (short)0, c, false, false);
}
static __device__ inline float sigm(float x) { return 1.f / (1.f + __expf(-x)); }

// ---- grid-wide barrier (16 resident WGs) -----------------------------------
static __device__ inline void gridbar(unsigned* cnt, unsigned* gen) {
  __threadfence();              // make this thread's global writes visible
  __syncthreads();
  if (threadIdx.x == 0) {
    __builtin_amdgcn_s_cluster_barrier();   // fast path if dispatched as a
                                            // 16-WG cluster; S_NOP otherwise
    unsigned g = *(volatile unsigned*)gen;
    unsigned prev = atomicAdd(cnt, 1u);
    if (prev == NWG - 1u) {
      *(volatile unsigned*)cnt = 0u;
      __threadfence();
      atomicAdd(gen, 1u);
    } else {
      while (*(volatile unsigned*)gen == g) __builtin_amdgcn_s_sleep(2);
    }
    __threadfence();
  }
  __syncthreads();
}

// ---- pack weights/activations to bf16, init recurrent buffers --------------
__global__ __launch_bounds__(256) void prep_pack(
    const float* __restrict__ enc, const float* __restrict__ h0,
    const float* __restrict__ Wq, const float* __restrict__ Wk,
    const float* __restrict__ Wout,
    const float* __restrict__ Wih0, const float* __restrict__ Whh0,
    const float* __restrict__ Wih1, const float* __restrict__ Whh1,
    __bf16* enc_bf, __bf16* Wq_bf, __bf16* Wk_bf, __bf16* Wout_bf,
    __bf16* WihE, __bf16* W0cat, __bf16* W1cat,
    __bf16* A0_0, __bf16* A1_0, __bf16* h1bf0) {
  const long N0 = (long)B_ * T_ * H_;   // enc        2097152
  const long N1 = (long)H_ * H_;        // Wq / Wk     262144
  const long N3 = (long)V_ * H_;        // Wout         16384
  const long N4 = (long)G_ * H_;        // WihE       1048576
  const long N5 = (long)G_ * K2_;       // W0cat/W1cat 2097152
  const long N7 = (long)B_ * H_;        // h-state init  8192
  const long total = N0 + 2 * N1 + N3 + N4 + 2 * N5 + N7;
  for (long i = blockIdx.x * (long)blockDim.x + threadIdx.x; i < total;
       i += (long)gridDim.x * blockDim.x) {
    long j = i;
    if (j < N0) { enc_bf[j] = (__bf16)enc[j]; continue; } j -= N0;
    if (j < N1) { Wq_bf[j] = (__bf16)Wq[j]; continue; }   j -= N1;
    if (j < N1) { Wk_bf[j] = (__bf16)Wk[j]; continue; }   j -= N1;
    if (j < N3) { Wout_bf[j] = (__bf16)Wout[j]; continue; } j -= N3;
    if (j < N4) { long n = j >> 9, k = j & 511;
                  WihE[j] = (__bf16)Wih0[n * K2_ + k]; continue; } j -= N4;
    if (j < N5) { long n = j >> 10, k = j & 1023;
                  W0cat[j] = (__bf16)(k < 512 ? Wih0[n * K2_ + 512 + k]
                                              : Whh0[n * H_ + (k - 512)]);
                  continue; } j -= N5;
    if (j < N5) { long n = j >> 10, k = j & 1023;
                  W1cat[j] = (__bf16)(k < 512 ? Wih1[n * H_ + k]
                                              : Whh1[n * H_ + (k - 512)]);
                  continue; } j -= N5;
    { long b = j >> 9, k = j & 511;
      __bf16 h0v = (__bf16)h0[b * H_ + k];
      __bf16 h1v = (__bf16)h0[(long)B_ * H_ + b * H_ + k];
      A0_0[b * K2_ + 512 + k] = h0v;   // layer0 h  -> A0 parity0 h-half
      A1_0[b * K2_ + 512 + k] = h1v;   // layer1 h  -> A1 parity0 h1-half
      h1bf0[b * H_ + k] = h1v; }
  }
}

// ---- big parallel GEMM: Out[4096,N] = enc_bf[4096,512] @ W^T + bias --------
__global__ __launch_bounds__(256) void gemm_enc_w(
    const __bf16* __restrict__ Abf, const __bf16* __restrict__ Wbf,
    const float* __restrict__ bias0, const float* __restrict__ bias1,
    float* __restrict__ Out, int N, int Mtiles) {
  const int wave = threadIdx.x >> 5, lane = threadIdx.x & 31;
  const int ntiles = N >> 4;
  const int tile = blockIdx.x * 8 + wave;
  if (tile >= Mtiles * ntiles) return;
  const int mt = tile / ntiles, nt = tile - mt * ntiles;
  const int m16 = lane & 15, hi = lane >> 4;
  const int khalf = hi * 16, ks2 = hi * 16, mo = hi * 8;
  const __bf16* Arow = Abf + (long)(mt * 16 + m16) * H_;
  const __bf16* Wrow = Wbf + (long)(nt * 16 + m16) * H_;
  v8f acc = {0, 0, 0, 0, 0, 0, 0, 0};
  for (int k = 0; k < H_; k += 32)
    acc = wmma_bf16(loadA_frag(Arow + k, khalf), loadB_frag(Wrow + k + ks2), acc);
  const int ncol = nt * 16 + m16;
  float bb = bias0[ncol] + (bias1 ? bias1[ncol] : 0.f);
#pragma unroll
  for (int j = 0; j < 8; ++j)
    Out[(long)(mt * 16 + mo + j) * N + ncol] = acc[j] + bb;
}

// ---- persistent decoder scan ----------------------------------------------
__global__ __launch_bounds__(256) void decoder_persistent(
    const float* __restrict__ enc, const float* __restrict__ c0in,
    const float* __restrict__ bq, const float* __restrict__ vvec,
    const float* __restrict__ vb, const float* __restrict__ bih1,
    const float* __restrict__ bhh1, const float* __restrict__ bout,
    const __bf16* __restrict__ Wq_bf, const __bf16* __restrict__ W0cat,
    const __bf16* __restrict__ W1cat, const __bf16* __restrict__ Wout_bf,
    const float* __restrict__ Kproj, const float* __restrict__ Xe,
    float* __restrict__ qproj, float* __restrict__ scores,
    __bf16* __restrict__ A0, __bf16* __restrict__ A1,
    __bf16* __restrict__ h1bf, unsigned* bar, float* __restrict__ out) {
  const int wg = blockIdx.x;
  const int tid = threadIdx.x;
  const int wave = tid >> 5, lane = tid & 31;
  const int m16 = lane & 15, hi = lane >> 4;
  const int khalf = hi * 16, ks2 = hi * 16, mo = hi * 8;

  __shared__ float sG[B_ * 128];   // gates for this WG's 32 hidden units
  __shared__ float sRed[256];
  __shared__ float sW[T_];

  // cell state lives in registers: WG wg permanently owns units [32wg,32wg+32)
  float c0r[2], c1r[2];
#pragma unroll
  for (int q = 0; q < 2; ++q) {
    int idx = tid + q * 256, b = idx & 15, ul = idx >> 4;
    int u = wg * 32 + ul;
    c0r[q] = c0in[b * H_ + u];
    c1r[q] = c0in[B_ * H_ + b * H_ + u];
  }
  unsigned* cnt = bar;
  unsigned* gen = bar + 1;

  for (int t = 0; t < T_; ++t) {
    const int p = t & 1;
    const __bf16* h1p = h1bf + p * (B_ * H_);

    // -- A: qproj = h1 @ Wq^T + bq   (this WG: 32 cols, 2 waves, WMMA) ------
    if (wave < 2) {
      const int nbase = wg * 32 + wave * 16;
      const __bf16* Arow = h1p + m16 * H_;
      const __bf16* Wrow = Wq_bf + (long)(nbase + m16) * H_;
      v8f acc = {0, 0, 0, 0, 0, 0, 0, 0};
      for (int k = 0; k < H_; k += 32)
        acc = wmma_bf16(loadA_frag(Arow + k, khalf), loadB_frag(Wrow + k + ks2), acc);
      const int ncol = nbase + m16;
      const float bb = bq[ncol];
#pragma unroll
      for (int j = 0; j < 8; ++j) qproj[(mo + j) * H_ + ncol] = acc[j] + bb;
    }
    gridbar(cnt, gen);

    // -- B: scores[b=wg][tt=tid] = vb + sum_h v*tanh(qproj + Kproj) ---------
    {
      const float* qp = qproj + wg * H_;
      const float* kp = Kproj + (long)(wg * T_ + tid) * H_;
      float s = vb[0];
      for (int h = 0; h < H_; ++h) s += vvec[h] * tanhf(qp[h] + kp[h]);
      scores[wg * T_ + tid] = s;
    }
    gridbar(cnt, gen);

    // -- C: softmax over T + ctx (b = wg); ctx written bf16 into A0[p] ------
    {
      float s = scores[wg * T_ + tid];
      sRed[tid] = s; __syncthreads();
      for (int off = 128; off > 0; off >>= 1) {
        if (tid < off) sRed[tid] = fmaxf(sRed[tid], sRed[tid + off]);
        __syncthreads();
      }
      float mx = sRed[0]; __syncthreads();
      float e = __expf(s - mx);
      sRed[tid] = e; __syncthreads();
      for (int off = 128; off > 0; off >>= 1) {
        if (tid < off) sRed[tid] += sRed[tid + off];
        __syncthreads();
      }
      float inv = 1.f / sRed[0]; __syncthreads();
      sW[tid] = e * inv; __syncthreads();
      const float* encb = enc + (long)wg * T_ * H_;
      __bf16* a0 = A0 + p * (B_ * K2_) + wg * K2_;
#pragma unroll
      for (int q = 0; q < 2; ++q) {
        int h = tid + q * 256;
        float acc = 0.f;
        for (int tt = 0; tt < T_; ++tt) acc += sW[tt] * encb[tt * H_ + h];
        a0[h] = (__bf16)acc;
      }
    }
    gridbar(cnt, gen);

    // -- D: gates0 = [ctx|h0]@W0cat^T + Xe(t); LSTM0 pointwise --------------
    {
      const int gi = wave >> 1, sub = wave & 1;           // gate i/f/g/o
      const int ncol0 = gi * H_ + wg * 32 + sub * 16;
      const __bf16* Arow = A0 + p * (B_ * K2_) + m16 * K2_;
      const __bf16* Wrow = W0cat + (long)(ncol0 + m16) * K2_;
      v8f acc = {0, 0, 0, 0, 0, 0, 0, 0};
      for (int k = 0; k < K2_; k += 32)
        acc = wmma_bf16(loadA_frag(Arow + k, khalf), loadB_frag(Wrow + k + ks2), acc);
      const int ncol = ncol0 + m16;
      const int lcol = gi * 32 + sub * 16 + m16;
#pragma unroll
      for (int j = 0; j < 8; ++j) {
        int b = mo + j;
        sG[b * 128 + lcol] = acc[j] + Xe[((long)b * T_ + t) * G_ + ncol];
      }
    }
    __syncthreads();
    {
      __bf16* a1 = A1 + p * (B_ * K2_);            // h0n -> A1[p] (read in E)
      __bf16* a0n = A0 + (1 - p) * (B_ * K2_);     // h0n -> A0[1-p] (step t+1)
#pragma unroll
      for (int q = 0; q < 2; ++q) {
        int idx = tid + q * 256, b = idx & 15, ul = idx >> 4;
        float gI = sG[b * 128 + ul];
        float gF = sG[b * 128 + 32 + ul];
        float gG = sG[b * 128 + 64 + ul];
        float gO = sG[b * 128 + 96 + ul];
        float c = sigm(gF) * c0r[q] + sigm(gI) * tanhf(gG);
        c0r[q] = c;
        float hn = sigm(gO) * tanhf(c);
        int u = wg * 32 + ul;
        __bf16 hb = (__bf16)hn;
        a1[b * K2_ + u] = hb;
        a0n[b * K2_ + 512 + u] = hb;
      }
    }
    gridbar(cnt, gen);

    // -- E: gates1 = [h0n|h1]@W1cat^T + biases; LSTM1 pointwise -------------
    {
      const int gi = wave >> 1, sub = wave & 1;
      const int ncol0 = gi * H_ + wg * 32 + sub * 16;
      const __bf16* Arow = A1 + p * (B_ * K2_) + m16 * K2_;
      const __bf16* Wrow = W1cat + (long)(ncol0 + m16) * K2_;
      v8f acc = {0, 0, 0, 0, 0, 0, 0, 0};
      for (int k = 0; k < K2_; k += 32)
        acc = wmma_bf16(loadA_frag(Arow + k, khalf), loadB_frag(Wrow + k + ks2), acc);
      const int ncol = ncol0 + m16;
      const int lcol = gi * 32 + sub * 16 + m16;
      const float bb = bih1[ncol] + bhh1[ncol];
#pragma unroll
      for (int j = 0; j < 8; ++j) sG[(mo + j) * 128 + lcol] = acc[j] + bb;
    }
    __syncthreads();
    {
      __bf16* a1n = A1 + (1 - p) * (B_ * K2_);     // h1n -> A1[1-p] (t+1)
      __bf16* h1n = h1bf + (1 - p) * (B_ * H_);    // h1n -> qproj/logits src
#pragma unroll
      for (int q = 0; q < 2; ++q) {
        int idx = tid + q * 256, b = idx & 15, ul = idx >> 4;
        float gI = sG[b * 128 + ul];
        float gF = sG[b * 128 + 32 + ul];
        float gG = sG[b * 128 + 64 + ul];
        float gO = sG[b * 128 + 96 + ul];
        float c = sigm(gF) * c1r[q] + sigm(gI) * tanhf(gG);
        c1r[q] = c;
        float hn = sigm(gO) * tanhf(c);
        int u = wg * 32 + ul;
        __bf16 hb = (__bf16)hn;
        a1n[b * K2_ + 512 + u] = hb;
        h1n[b * H_ + u] = hb;
      }
    }
    gridbar(cnt, gen);

    // -- F: logits = h1n @ Wout^T + bout (WG0, 2 waves; overlaps next A) ----
    if (wg == 0 && wave < 2) {
      const __bf16* h1n = h1bf + (1 - p) * (B_ * H_);
      const int nbase = wave * 16;
      const __bf16* Arow = h1n + m16 * H_;
      const __bf16* Wrow = Wout_bf + (long)(nbase + m16) * H_;
      v8f acc = {0, 0, 0, 0, 0, 0, 0, 0};
      for (int k = 0; k < H_; k += 32)
        acc = wmma_bf16(loadA_frag(Arow + k, khalf), loadB_frag(Wrow + k + ks2), acc);
      const int ncol = nbase + m16;
      const float bb = bout[ncol];
#pragma unroll
      for (int j = 0; j < 8; ++j)
        out[((long)(mo + j) * T_ + t) * V_ + ncol] = acc[j] + bb;
    }
    // no barrier needed: next phase A only reads h1bf (already fenced by the
    // post-E barrier); the post-A barrier orders qproj for phase B.
  }
}

// ---------------------------------------------------------------------------
extern "C" void kernel_launch(void* const* d_in, const int* in_sizes, int n_in,
                              void* d_out, int out_size, void* d_ws,
                              size_t ws_size, hipStream_t stream) {
  (void)in_sizes; (void)n_in; (void)out_size; (void)ws_size;
  const float* enc  = (const float*)d_in[0];
  const float* h0   = (const float*)d_in[1];
  const float* c0   = (const float*)d_in[2];
  /* d_in[3] audio_lengths: unused by the reference math */
  const float* Wq   = (const float*)d_in[4];
  const float* bq   = (const float*)d_in[5];
  const float* Wk   = (const float*)d_in[6];
  const float* bk   = (const float*)d_in[7];
  const float* v    = (const float*)d_in[8];
  const float* vb   = (const float*)d_in[9];
  const float* Wih0 = (const float*)d_in[10];
  const float* bih0 = (const float*)d_in[11];
  const float* Whh0 = (const float*)d_in[12];
  const float* bhh0 = (const float*)d_in[13];
  const float* Wih1 = (const float*)d_in[14];
  const float* bih1 = (const float*)d_in[15];
  const float* Whh1 = (const float*)d_in[16];
  const float* bhh1 = (const float*)d_in[17];
  const float* Wout = (const float*)d_in[18];
  const float* bout = (const float*)d_in[19];
  float* out = (float*)d_out;

  // workspace carve (~58 MB total, all 256B aligned)
  char* w = (char*)d_ws;
  size_t o = 0;
  auto carve = [&](size_t bytes) {
    char* r = w + o;
    o = (o + bytes + 255) & ~(size_t)255;
    return r;
  };
  unsigned* bar  = (unsigned*)carve(256);
  __bf16* enc_bf = (__bf16*)carve((size_t)B_ * T_ * H_ * 2);
  __bf16* Wq_bf  = (__bf16*)carve((size_t)H_ * H_ * 2);
  __bf16* Wk_bf  = (__bf16*)carve((size_t)H_ * H_ * 2);
  __bf16* Wout_bf= (__bf16*)carve((size_t)V_ * H_ * 2);
  __bf16* WihE   = (__bf16*)carve((size_t)G_ * H_ * 2);
  __bf16* W0cat  = (__bf16*)carve((size_t)G_ * K2_ * 2);
  __bf16* W1cat  = (__bf16*)carve((size_t)G_ * K2_ * 2);
  float*  Kproj  = (float*)carve((size_t)B_ * T_ * H_ * 4);
  float*  Xe     = (float*)carve((size_t)B_ * T_ * G_ * 4);
  float*  qproj  = (float*)carve((size_t)B_ * H_ * 4);
  float*  scores = (float*)carve((size_t)B_ * T_ * 4);
  __bf16* A0     = (__bf16*)carve((size_t)2 * B_ * K2_ * 2);
  __bf16* A1     = (__bf16*)carve((size_t)2 * B_ * K2_ * 2);
  __bf16* h1bf   = (__bf16*)carve((size_t)2 * B_ * H_ * 2);

  hipMemsetAsync(bar, 0, 256, stream);   // deterministic barrier state

  prep_pack<<<2048, 256, 0, stream>>>(enc, h0, Wq, Wk, Wout, Wih0, Whh0,
                                      Wih1, Whh1, enc_bf, Wq_bf, Wk_bf,
                                      Wout_bf, WihE, W0cat, W1cat,
                                      A0, A1, h1bf);
  // Kproj: 256x32 tiles / 8 waves per block
  gemm_enc_w<<<1024, 256, 0, stream>>>(enc_bf, Wk_bf, bk, nullptr, Kproj,
                                       H_, 256);
  // Xe: 256x128 tiles / 8 waves per block
  gemm_enc_w<<<4096, 256, 0, stream>>>(enc_bf, WihE, bih0, bhh0, Xe,
                                       G_, 256);
  decoder_persistent<<<NWG, 256, 0, stream>>>(
      enc, c0, bq, v, vb, bih1, bhh1, bout, Wq_bf, W0cat, W1cat, Wout_bf,
      Kproj, Xe, qproj, scores, A0, A1, h1bf, bar, out);
}
